// SelfAttention_65481071399119
// MI455X (gfx1250) — matrix operand order repaired and checked by reading, not compiled, not run on hardware
//
#include <hip/hip_runtime.h>

typedef __attribute__((ext_vector_type(16))) __bf16 v16bf;
typedef __attribute__((ext_vector_type(8)))  __bf16 v8bf;
typedef __attribute__((ext_vector_type(8)))  float  v8f;

#define BATCH 32
#define CH    4096
#define C8    512
#define HW    512   // 16*32 == C8

// ---- GEMM tiling ----
#define BM 128
#define BN 128
#define BK 32
#define LDS_STRIDE 40                 // elements; 80B row stride -> conflict-free b128
#define SLAB_ELEMS (BM * LDS_STRIDE)  // 5120 elements
#define SLAB_BYTES (SLAB_ELEMS * 2)   // 10240 B per buffer

#define USE_ASYNC 1

#if USE_ASYNC
__device__ __forceinline__ void async_b128(unsigned lds_byte_off, const __bf16* g) {
  asm volatile("global_load_async_to_lds_b128 %0, %1, off"
               :: "v"(lds_byte_off), "v"(g) : "memory");
}
__device__ __forceinline__ void wait_async0() {
  asm volatile("s_wait_asynccnt 0x0" ::: "memory");
}
#endif

// ---------------- cast kernels ----------------
__global__ void cast_f32_to_bf16(const float* __restrict__ src,
                                 __bf16* __restrict__ dst, int n) {
  int i = blockIdx.x * blockDim.x + threadIdx.x;
  if (i < n) dst[i] = (__bf16)src[i];
}

// x (B, C, L) f32 -> xbt (B, L, C) bf16  (GEMM B-operand becomes K-contiguous)
__global__ void transpose_cast_x(const float* __restrict__ x,
                                 __bf16* __restrict__ xbt) {
  long i = (long)blockIdx.x * blockDim.x + threadIdx.x;
  const long total = (long)BATCH * CH * HW;
  if (i >= total) return;
  long b = i / ((long)CH * HW);
  long r = i - b * (long)CH * HW;
  long c = r / HW;
  long l = r - c * HW;
  xbt[b * (long)HW * CH + l * CH + c] = (__bf16)x[i];
}

// ------------- LDS-staged WMMA GEMM:  C[m,n] = sum_k A[m,k]*B[n,k] -------------
// 256 threads, 8 waves; block tile 128x128, wave tile 64x32 (4x2 WMMA frags).
// MODE 0: Cb[m,n] = bf16(acc + bias[m])
// MODE 1: Cf[m,n] = acc
// MODE 2: Cf[m,n] = gamma*acc + X[m,n]
template<int MODE>
__global__ __launch_bounds__(256)
void wmma_gemm_abT(const __bf16* __restrict__ A, long sA,
                   const __bf16* __restrict__ Bm, long sB,
                   const float*  __restrict__ bias,
                   __bf16* __restrict__ Cb,
                   float*  __restrict__ Cf,
                   const float* __restrict__ X,
                   const float* __restrict__ gamma,
                   long sC, int M, int N, int K) {
  __shared__ __bf16 As[2][SLAB_ELEMS];
  __shared__ __bf16 Bs[2][SLAB_ELEMS];

  const int tid  = threadIdx.x;
  const int lane = tid & 31;
  const int wave = tid >> 5;
  const int wm   = wave >> 2;   // 0..1  (64-row band)
  const int wn   = wave & 3;    // 0..3  (32-col band)
  const int half = lane >> 4;
  const int l16  = lane & 15;

  const int nt = N / BN;
  const int tm = (blockIdx.x / nt) * BM;
  const int tn = (blockIdx.x % nt) * BN;
  const int b  = blockIdx.y;

  const __bf16* Ab = A  + (long)b * sA;
  const __bf16* Bb = Bm + (long)b * sB;

  // slab = 128 rows x 32 bf16 = 512 x 16B chunks; 2 chunks per thread
  int rowS[2], colS[2];
#pragma unroll
  for (int i = 0; i < 2; ++i) {
    const int chunk = tid + i * 256;
    rowS[i] = chunk >> 2;
    colS[i] = (chunk & 3) * 8;  // element offset within row
  }
  const unsigned asBase = (unsigned)(size_t)&As[0][0];
  const unsigned bsBase = (unsigned)(size_t)&Bs[0][0];

  v8f acc[4][2];
#pragma unroll
  for (int f = 0; f < 4; ++f)
#pragma unroll
    for (int g = 0; g < 2; ++g) acc[f][g] = (v8f){};

  const int NK = K / BK;

#if USE_ASYNC
  // prologue: stream slab 0 into buffer 0
#pragma unroll
  for (int i = 0; i < 2; ++i) {
    async_b128(asBase + rowS[i] * (LDS_STRIDE * 2) + colS[i] * 2,
               Ab + (long)rowS[i] * K + colS[i]);
    async_b128(bsBase + rowS[i] * (LDS_STRIDE * 2) + colS[i] * 2,
               Bb + (long)rowS[i] * K + colS[i]);
  }
#endif

  for (int kt = 0; kt < NK; ++kt) {
    const int buf = kt & 1;
#if USE_ASYNC
    wait_async0();        // my slab-kt writes to LDS have landed
    __syncthreads();      // everyone's slab kt visible; buf^1 readers drained
    if (kt + 1 < NK) {    // stream slab kt+1 into the other buffer during compute
      const int k0 = (kt + 1) * BK;
      const unsigned o = (unsigned)((buf ^ 1) * SLAB_BYTES);
#pragma unroll
      for (int i = 0; i < 2; ++i) {
        async_b128(asBase + o + rowS[i] * (LDS_STRIDE * 2) + colS[i] * 2,
                   Ab + (long)rowS[i] * K + k0 + colS[i]);
        async_b128(bsBase + o + rowS[i] * (LDS_STRIDE * 2) + colS[i] * 2,
                   Bb + (long)rowS[i] * K + k0 + colS[i]);
      }
    }
#else
    __syncthreads();      // previous readers of this buffer are done
    {
      const int k0 = kt * BK;
#pragma unroll
      for (int i = 0; i < 2; ++i) {
        *(v8bf*)&As[buf][rowS[i] * LDS_STRIDE + colS[i]] =
            *(const v8bf*)(Ab + (long)rowS[i] * K + k0 + colS[i]);
        *(v8bf*)&Bs[buf][rowS[i] * LDS_STRIDE + colS[i]] =
            *(const v8bf*)(Bb + (long)rowS[i] * K + k0 + colS[i]);
      }
    }
    __syncthreads();
#endif

    // fragments from LDS (layouts per ISA 7.12.2) + 8 WMMAs
    v16bf af[4], bf[2];
#pragma unroll
    for (int f = 0; f < 4; ++f) {
      const __bf16* pa = &As[buf][(wm * 64 + f * 16 + l16) * LDS_STRIDE + half * 8];
      v8bf lo = *(const v8bf*)pa;
      v8bf hi = *(const v8bf*)(pa + 16);
      af[f] = __builtin_shufflevector(lo, hi, 0,1,2,3,4,5,6,7,8,9,10,11,12,13,14,15);
    }
#pragma unroll
    for (int g = 0; g < 2; ++g) {
      const __bf16* pb = &Bs[buf][(wn * 32 + g * 16 + l16) * LDS_STRIDE + half * 8];
      v8bf lo = *(const v8bf*)pb;
      v8bf hi = *(const v8bf*)(pb + 16);
      bf[g] = __builtin_shufflevector(lo, hi, 0,1,2,3,4,5,6,7,8,9,10,11,12,13,14,15);
    }
#pragma unroll
    for (int f = 0; f < 4; ++f)
#pragma unroll
      for (int g = 0; g < 2; ++g)
        acc[f][g] = __builtin_amdgcn_wmma_f32_16x16x32_bf16(
            false, af[f], false, bf[g], (short)0, acc[f][g], false, false);
  }

  // epilogue: vgpr r of each 16x16 frag -> (M = base + half*8 + r, N = base + l16)
#pragma unroll
  for (int f = 0; f < 4; ++f) {
    const int m = tm + wm * 64 + f * 16 + half * 8;
#pragma unroll
    for (int r = 0; r < 8; ++r) {
#pragma unroll
      for (int g = 0; g < 2; ++g) {
        const int n = tn + wn * 32 + g * 16 + l16;
        const long idx = (long)b * sC + (long)(m + r) * N + n;
        const float v = acc[f][g][r];
        if (MODE == 0) {
          Cb[idx] = (__bf16)(v + bias[m + r]);
        } else if (MODE == 1) {
          Cf[idx] = v;
        } else {
          Cf[idx] = gamma[0] * v + X[idx];
        }
      }
    }
  }
}

// ---------------- softmax over rows of 512, one wave per row ----------------
__global__ void softmax_rows(const float* __restrict__ E,
                             __bf16* __restrict__ Attn, int rows) {
  const int row  = blockIdx.x * (blockDim.x >> 5) + (threadIdx.x >> 5);
  const int lane = threadIdx.x & 31;
  if (row >= rows) return;
  const float* e = E + (long)row * HW;
  float vals[16];
  float m = -3.4e38f;
#pragma unroll
  for (int j = 0; j < 16; ++j) { vals[j] = e[lane + 32 * j]; m = fmaxf(m, vals[j]); }
#pragma unroll
  for (int off = 16; off > 0; off >>= 1) m = fmaxf(m, __shfl_xor(m, off, 32));
  float s = 0.f;
#pragma unroll
  for (int j = 0; j < 16; ++j) { vals[j] = __expf(vals[j] - m); s += vals[j]; }
#pragma unroll
  for (int off = 16; off > 0; off >>= 1) s += __shfl_xor(s, off, 32);
  const float inv = 1.0f / s;
  __bf16* a = Attn + (long)row * HW;
#pragma unroll
  for (int j = 0; j < 16; ++j) a[lane + 32 * j] = (__bf16)(vals[j] * inv);
}

// ---------------- launch ----------------
extern "C" void kernel_launch(void* const* d_in, const int* in_sizes, int n_in,
                              void* d_out, int out_size, void* d_ws, size_t ws_size,
                              hipStream_t stream) {
  const float* x     = (const float*)d_in[0];
  const float* Wq    = (const float*)d_in[1];
  const float* bq    = (const float*)d_in[2];
  const float* Wk    = (const float*)d_in[3];
  const float* bk    = (const float*)d_in[4];
  const float* Wv    = (const float*)d_in[5];
  const float* bv    = (const float*)d_in[6];
  const float* gamma = (const float*)d_in[7];
  float* out = (float*)d_out;

  char* p = (char*)d_ws;
  __bf16* xbt = (__bf16*)p; p += (long)BATCH * HW * CH * 2;   // (b,l,c)
  __bf16* Wqb = (__bf16*)p; p += (long)C8 * CH * 2;
  __bf16* Wkb = (__bf16*)p; p += (long)C8 * CH * 2;
  __bf16* Wvb = (__bf16*)p; p += (long)CH * CH * 2;
  __bf16* qb  = (__bf16*)p; p += (long)BATCH * C8 * HW * 2;   // (b,o,l)
  __bf16* kb  = (__bf16*)p; p += (long)BATCH * C8 * HW * 2;
  __bf16* vb  = (__bf16*)p; p += (long)BATCH * CH * HW * 2;   // (b,c,l)
  float*  en  = (float*)p;  p += (long)BATCH * C8 * C8 * 4;
  __bf16* attn = qb;  // q is dead after the energy GEMM; reuse (same size)

  const int T = 256;
  transpose_cast_x<<<(BATCH * CH * HW + T - 1) / T, T, 0, stream>>>(x, xbt);
  cast_f32_to_bf16<<<(C8 * CH + T - 1) / T, T, 0, stream>>>(Wq, Wqb, C8 * CH);
  cast_f32_to_bf16<<<(C8 * CH + T - 1) / T, T, 0, stream>>>(Wk, Wkb, C8 * CH);
  cast_f32_to_bf16<<<(CH * CH + T - 1) / T, T, 0, stream>>>(Wv, Wvb, CH * CH);

  // projections (weights shared across batch -> stride 0)
  {
    dim3 g((C8 / BM) * (HW / BN), BATCH);
    wmma_gemm_abT<0><<<g, T, 0, stream>>>(Wqb, 0, xbt, (long)HW * CH, bq,
                                          qb, nullptr, nullptr, nullptr,
                                          (long)C8 * HW, C8, HW, CH);
    wmma_gemm_abT<0><<<g, T, 0, stream>>>(Wkb, 0, xbt, (long)HW * CH, bk,
                                          kb, nullptr, nullptr, nullptr,
                                          (long)C8 * HW, C8, HW, CH);
  }
  {
    dim3 g((CH / BM) * (HW / BN), BATCH);
    wmma_gemm_abT<0><<<g, T, 0, stream>>>(Wvb, 0, xbt, (long)HW * CH, bv,
                                          vb, nullptr, nullptr, nullptr,
                                          (long)CH * HW, CH, HW, CH);
  }

  // energy = q . k^T  (f32)
  {
    dim3 g((C8 / BM) * (C8 / BN), BATCH);
    wmma_gemm_abT<1><<<g, T, 0, stream>>>(qb, (long)C8 * HW, kb, (long)C8 * HW,
                                          nullptr, nullptr, en, nullptr, nullptr,
                                          (long)C8 * C8, C8, C8, HW);
  }

  // softmax rows -> attn (bf16)
  softmax_rows<<<(BATCH * C8 + 7) / 8, T, 0, stream>>>(en, attn, BATCH * C8);

  // out = gamma * (v . attn^T) + x
  {
    dim3 g((CH / BM) * (HW / BN), BATCH);
    wmma_gemm_abT<2><<<g, T, 0, stream>>>(vb, (long)CH * HW, attn, (long)C8 * C8,
                                          nullptr, nullptr, out, x, gamma,
                                          (long)CH * HW, CH, HW, C8);
  }
}
